// CustomLSTMModel_4509715661353
// MI455X (gfx1250) — matrix operand order repaired and checked by reading, not compiled, not run on hardware
//
#include <hip/hip_runtime.h>
#include <hip/hip_bf16.h>

// ---------------------------------------------------------------------------
// LSTM text classifier for MI455X (gfx1250), wave32 + WMMA f16->f32.
// Per timestep t (fused input projection):
//   gates = [x_t | h] @ [W_ih | W_hh]^T + (b_ih + b_hh),  K = 320(pad) + 512
// v_wmma_f32_16x16x32_f16 with register double-buffered A/B fragments,
// M=32 per wave (two row tiles share the 4 gate B-fragments), and fully
// vectorized LDS staging (float4 embed gather, b128 h-panel copy).
// ---------------------------------------------------------------------------

typedef __attribute__((ext_vector_type(16))) _Float16 v16h;
typedef __attribute__((ext_vector_type(8)))  _Float16 v8h;
typedef __attribute__((ext_vector_type(4)))  _Float16 v4h;
typedef __attribute__((ext_vector_type(8)))  float    v8f;

#define BATCH 256
#define SEQ   512
#define EMBD  300
#define EPAD  320               // E padded to multiple of 32 (K granularity)
#define HID   512
#define KTOT  (EPAD + HID)      // 832 combined K
#define G4H   (4 * HID)         // 2048 gate columns
#define MROWS 32                // rows per block (two 16-row WMMA tiles/wave)

union V16U { v16h v; v8h h[2]; };

__device__ __forceinline__ float fast_sigmoid(float x) {
  return 1.0f / (1.0f + __expf(-x));
}
__device__ __forceinline__ float fast_tanh(float x) {
  float e = __expf(2.0f * x);
  return (e - 1.0f) / (e + 1.0f);
}

// --- Pack [W_ih | 0-pad | W_hh] into one f16 matrix [2048, 832]; fuse biases.
__global__ void prep_weights(const float* __restrict__ W_ih,
                             const float* __restrict__ W_hh,
                             const float* __restrict__ b_ih,
                             const float* __restrict__ b_hh,
                             _Float16* __restrict__ Wcat,
                             float* __restrict__ bias) {
  const int idx = blockIdx.x * blockDim.x + threadIdx.x;
  const int total = G4H * KTOT;
  for (int i = idx; i < total; i += gridDim.x * blockDim.x) {
    const int n = i / KTOT;
    const int k = i % KTOT;
    float v;
    if (k < EMBD)       v = W_ih[n * EMBD + k];
    else if (k < EPAD)  v = 0.0f;
    else                v = W_hh[n * HID + (k - EPAD)];
    Wcat[i] = (_Float16)v;
  }
  if (idx < G4H) bias[idx] = b_ih[idx] + b_hh[idx];
}

// --- Zero h (both buffers) and cell state c (workspace is poisoned 0xAA).
__global__ void init_state(_Float16* __restrict__ h_a,
                           _Float16* __restrict__ h_b,
                           float* __restrict__ c) {
  const int idx = blockIdx.x * blockDim.x + threadIdx.x;
  const int n = BATCH * HID;
  for (int i = idx; i < n; i += gridDim.x * blockDim.x) {
    h_a[i] = (_Float16)0.0f;
    h_b[i] = (_Float16)0.0f;
    c[i]   = 0.0f;
  }
}

// --- One LSTM timestep.
// grid = (8 col-blocks of 64 h-cols, 8 row-tiles of 32 batch rows), 128 thr.
// Each wave: 32x16 output tile x 4 gates; double-buffered fragments.
__global__ __launch_bounds__(128)
void lstm_step(const int* __restrict__ tokens, const float* __restrict__ emb,
               const _Float16* __restrict__ Wcat, const float* __restrict__ bias,
               const _Float16* __restrict__ h_in, _Float16* __restrict__ h_out,
               float* __restrict__ c, int t) {
  __shared__ _Float16 sA[MROWS * KTOT];   // 53,248 B: [x_t(f16,pad) | h(f16)]
  __shared__ int sTok[MROWS];

  const int rb  = blockIdx.y * MROWS;     // batch-row base
  const int tid = threadIdx.x;

  if (tid < MROWS) sTok[tid] = tokens[(rb + tid) * SEQ + t];
  __syncthreads();

  // Stage embedded tokens, vectorized: 80 float4-chunks per row.
  // Chunks >= 75 cover the 300->320 zero pad; tok==0 rows are all zero.
  for (int i = tid; i < MROWS * (EPAD / 4); i += 128) {
    const int r = i / (EPAD / 4);
    const int q = i % (EPAD / 4);
    const int tok = sTok[r];
    v4h hv;
    if (q < EMBD / 4 && tok != 0) {
      const float4 f = *(const float4*)(emb + (size_t)tok * EMBD + 4 * q);
      hv = (v4h){(_Float16)f.x, (_Float16)f.y, (_Float16)f.z, (_Float16)f.w};
    } else {
      hv = (v4h){(_Float16)0.0f, (_Float16)0.0f, (_Float16)0.0f, (_Float16)0.0f};
    }
    *(v4h*)(&sA[r * KTOT + 4 * q]) = hv;   // 8-B aligned store
  }
  // Stage previous hidden state as 16-B copies (both sides 16-B aligned).
  for (int i = tid; i < MROWS * (HID / 8); i += 128) {
    const int r = i / (HID / 8);
    const int q = i % (HID / 8);
    *(v8h*)(&sA[r * KTOT + EPAD + 8 * q]) =
        *(const v8h*)(h_in + (size_t)(rb + r) * HID + 8 * q);
  }
  __syncthreads();

  const int wave  = tid >> 5;
  const int lane  = tid & 31;
  const int n0    = blockIdx.x * 64 + wave * 16;  // h-column base of tile
  const int bcol  = lane & 15;                    // B-matrix column (N)
  const int khalf = (lane < 16) ? 0 : 8;          // B lanes: K 0-7 vs 8-15
  const int alo   = (lane < 16) ? 0 : 8;          // A lanes: K 0-7 vs 8-15

  // A rows for the two 16-row tiles of this wave.
  const _Float16* sArow0 = &sA[(lane & 15) * KTOT + alo];
  const _Float16* sArow1 = &sA[((lane & 15) + 16) * KTOT + alo];
  // B base: column n0+bcol of gate 0; gates stride by HID rows of Wcat.
  const _Float16* wbase = Wcat + (size_t)(n0 + bcol) * KTOT + khalf;

  v8f acc[8] = {};   // [tile0: i,f,g,o | tile1: i,f,g,o]

  V16U aq[2][2];     // [buf][row tile]
  V16U bq[2][4];     // [buf][gate]

  auto load_frags = [&](int buf, int kb) {
    aq[buf][0].h[0] = *(const v8h*)(sArow0 + kb);
    aq[buf][0].h[1] = *(const v8h*)(sArow0 + kb + 16);
    aq[buf][1].h[0] = *(const v8h*)(sArow1 + kb);
    aq[buf][1].h[1] = *(const v8h*)(sArow1 + kb + 16);
#pragma unroll
    for (int g = 0; g < 4; ++g) {
      const _Float16* wp = wbase + (size_t)g * HID * KTOT + kb;
      bq[buf][g].h[0] = *(const v8h*)(wp);
      bq[buf][g].h[1] = *(const v8h*)(wp + 16);
    }
  };

  load_frags(0, 0);
  int buf = 0;
  for (int kb = 0; kb < KTOT; kb += 32) {
    if (kb + 32 < KTOT) load_frags(buf ^ 1, kb + 32);  // prefetch next chunk
#pragma unroll
    for (int g = 0; g < 4; ++g)
      acc[g] = __builtin_amdgcn_wmma_f32_16x16x32_f16(
          false, aq[buf][0].v, false, bq[buf][g].v, (short)0, acc[g],
          false, false);
#pragma unroll
    for (int g = 0; g < 4; ++g)
      acc[4 + g] = __builtin_amdgcn_wmma_f32_16x16x32_f16(
          false, aq[buf][1].v, false, bq[buf][g].v, (short)0, acc[4 + g],
          false, false);
    buf ^= 1;
  }

  // Epilogue: LSTM cell update for this wave's 32x16 tile.
  const int col = n0 + bcol;
  const float bi = bias[0 * HID + col];
  const float bf = bias[1 * HID + col];
  const float bg = bias[2 * HID + col];
  const float bo = bias[3 * HID + col];
  const int mbase = (lane < 16) ? 0 : 8;           // D rows per half-wave
#pragma unroll
  for (int tile = 0; tile < 2; ++tile) {
#pragma unroll
    for (int r = 0; r < 8; ++r) {
      const int m = rb + tile * 16 + mbase + r;
      const float ig = fast_sigmoid(acc[tile * 4 + 0][r] + bi);
      const float fg = fast_sigmoid(acc[tile * 4 + 1][r] + bf);
      const float gg = fast_tanh   (acc[tile * 4 + 2][r] + bg);
      const float og = fast_sigmoid(acc[tile * 4 + 3][r] + bo);
      const size_t idx = (size_t)m * HID + col;
      const float cn = fg * c[idx] + ig * gg;
      c[idx] = cn;
      h_out[idx] = (_Float16)(og * fast_tanh(cn));
    }
  }
}

// --- Final FC: logits[b,o] = h_T[b,:] . W_fc[o,:] + b_fc[o]  (tiny: 256x7)
__global__ void fc_kernel(const _Float16* __restrict__ h,
                          const float* __restrict__ W_fc,
                          const float* __restrict__ b_fc,
                          float* __restrict__ out) {
  const int gid = blockIdx.x * blockDim.x + threadIdx.x;
  if (gid >= BATCH * 7) return;
  const int b = gid / 7, o = gid % 7;
  float s = 0.0f;
  for (int k = 0; k < HID; ++k)
    s += (float)h[b * HID + k] * W_fc[o * HID + k];
  out[gid] = s + b_fc[o];
}

extern "C" void kernel_launch(void* const* d_in, const int* in_sizes, int n_in,
                              void* d_out, int out_size, void* d_ws,
                              size_t ws_size, hipStream_t stream) {
  (void)in_sizes; (void)n_in; (void)out_size; (void)ws_size;
  const int*   tokens = (const int*)d_in[0];
  const float* emb    = (const float*)d_in[1];
  const float* W_ih   = (const float*)d_in[2];
  const float* b_ih   = (const float*)d_in[3];
  const float* W_hh   = (const float*)d_in[4];
  const float* b_hh   = (const float*)d_in[5];
  const float* W_fc   = (const float*)d_in[6];
  const float* b_fc   = (const float*)d_in[7];
  float* out = (float*)d_out;

  // Workspace layout (≈4.5 MB total)
  char* ws = (char*)d_ws;
  _Float16* Wcat = (_Float16*)(ws + 0);          // 2048*832*2 = 3,407,872 B
  float*    bias = (float*)   (ws + 3407872);    // 2048*4     =     8,192 B
  _Float16* h_a  = (_Float16*)(ws + 3416064);    // 256*512*2  =   262,144 B
  _Float16* h_b  = (_Float16*)(ws + 3678208);    // 256*512*2  =   262,144 B
  float*    c    = (float*)   (ws + 3940352);    // 256*512*4  =   524,288 B

  prep_weights<<<512, 256, 0, stream>>>(W_ih, W_hh, b_ih, b_hh, Wcat, bias);
  init_state<<<512, 256, 0, stream>>>(h_a, h_b, c);

  dim3 grid(8, 8);   // 8 col-blocks x 8 row-tiles(32), 128 threads (4 waves)
  for (int t = 0; t < SEQ; ++t) {
    const _Float16* hin = (t & 1) ? h_b : h_a;
    _Float16*      hout = (t & 1) ? h_a : h_b;
    lstm_step<<<grid, 128, 0, stream>>>(tokens, emb, Wcat, bias, hin, hout, c, t);
  }
  // SEQ = 512 steps, last step index 511 (odd) writes h_a -> final hidden.
  fc_kernel<<<(BATCH * 7 + 255) / 256, 256, 0, stream>>>(h_a, W_fc, b_fc, out);
}
